// SpatioTemporalGCN_29532195127549
// MI455X (gfx1250) — matrix operand order repaired and checked by reading, not compiled, not run on hardware
//
#include <hip/hip_runtime.h>
#include <hip/hip_bf16.h>
#include <stdint.h>

// ---------------------------------------------------------------------------
// SpatioTemporal GCN for MI455X (gfx1250, wave32).
//  - 3x GCN layers; GEMMs via v_wmma_f32_16x16x32_f16 (f32 accum, f16 A/B;
//    activations are clipped to [-10,10] so f16 operands are safe).
//  - B (weights) staged in LDS per workgroup -> ds_load path, A streamed
//    from global with prefetch; independent DScnt/LOADcnt waits.
//  - Edge gather/scatter: warp-per-edge, coalesced, f32 global atomics
//    (bandwidth-bound part: ~2.5 GB irregular traffic ~= 110us at 23.3 TB/s).
// ---------------------------------------------------------------------------

typedef __attribute__((ext_vector_type(16))) _Float16 v16h;
typedef __attribute__((ext_vector_type(8)))  _Float16 v8h;
typedef __attribute__((ext_vector_type(8)))  float    v8f;

#define NNODES   100000
#define NSTATIC  16
#define NDYNFLAT 192        // SEQ*DYN = 24*8
#define NCOMB    208
#define KPAD0    224        // 208 padded to 7 * 32
#define NHID     64
#define BIGLIM   1.0e6f
#define BNEPS    1e-5f
#define MAXKSTEPS 7

__device__ __forceinline__ float cleanf(float x, float lim) {
    x = isnan(x) ? 0.0f : x;
    return fminf(fmaxf(x, -lim), lim);
}
__device__ __forceinline__ float clean_ew(float w) {
    w = isnan(w) ? 0.0f : w;
    return fminf(fmaxf(w, 0.0f), BIGLIM);
}

// ---------------------------------------------------------------------------
// small utility kernels
// ---------------------------------------------------------------------------
__global__ void k_zero_f32(float* p, int n) {
    int i = blockIdx.x * blockDim.x + threadIdx.x;
    if (i < n) p[i] = 0.0f;
}

__global__ void k_deg_init(float* deg, int n) {
    int i = blockIdx.x * blockDim.x + threadIdx.x;
    if (i < n) deg[i] = 1.0f;           // self-loop weight
}

__global__ void k_deg_scatter(const int* __restrict__ ei, const float* __restrict__ ew,
                              float* deg, long long E) {
    long long e = (long long)blockIdx.x * blockDim.x + threadIdx.x;
    if (e >= E) return;
    int col = ei[E + e];
    atomicAdd(&deg[col], clean_ew(ew[e]));
}

__global__ void k_dinv(float* deg, int n) {
    int i = blockIdx.x * blockDim.x + threadIdx.x;
    if (i < n) {
        float d = deg[i];
        deg[i] = (d > 0.0f) ? rsqrtf(d) : 0.0f;
    }
}

// ---------------------------------------------------------------------------
// BatchNorm statistics (biased variance, as in the reference)
// stats layout (floats): [0..F) sum, [224..224+F) sumsq, [448..) mean, [672..) rs
// ---------------------------------------------------------------------------
__global__ void k_stats_input(const float* __restrict__ xs, const float* __restrict__ xd,
                              float* __restrict__ stats, int n) {
    // thread t < 208 owns feature t; each block covers 256 rows
    int t = threadIdx.x;
    float s = 0.0f, q = 0.0f;
    if (t < NCOMB) {
        int r0 = blockIdx.x * 256;
        for (int i = 0; i < 256; ++i) {
            int r = r0 + i;
            if (r >= n) break;
            float v = (t < NSTATIC) ? xs[(size_t)r * NSTATIC + t]
                                    : xd[(size_t)r * NDYNFLAT + (t - NSTATIC)];
            v = cleanf(v, BIGLIM);
            s += v; q += v * v;
        }
        atomicAdd(&stats[t],        s);
        atomicAdd(&stats[224 + t],  q);
    }
}

__global__ void k_stats_hidden(const float* __restrict__ o, float* __restrict__ stats, int n) {
    // 256 threads = 4 rows in flight x 64 features; block covers 256 rows
    int f = threadIdx.x & 63;
    int rsub = threadIdx.x >> 6;
    float s = 0.0f, q = 0.0f;
    int r0 = blockIdx.x * 256;
    for (int i = 0; i < 64; ++i) {
        int r = r0 + i * 4 + rsub;
        if (r >= n) break;
        float v = cleanf(o[(size_t)r * NHID + f], 10.0f);
        s += v; q += v * v;
    }
    atomicAdd(&stats[f],       s);
    atomicAdd(&stats[224 + f], q);
}

__global__ void k_bn_finalize(float* stats, int F, float inv_n) {
    int t = threadIdx.x;
    if (t < F) {
        float m = stats[t] * inv_n;
        float v = stats[224 + t] * inv_n - m * m;
        stats[448 + t] = m;
        stats[672 + t] = rsqrtf(v + BNEPS);
    }
}

// ---------------------------------------------------------------------------
// layer-0 A-matrix fill: feats -> input BN -> clip(10) -> f16, K padded to 224
// ---------------------------------------------------------------------------
__global__ void k_fill_A0(const float* __restrict__ xs, const float* __restrict__ xd,
                          const float* __restrict__ stats,
                          const float* __restrict__ g, const float* __restrict__ b,
                          _Float16* __restrict__ A, int n) {
    long long idx = (long long)blockIdx.x * blockDim.x + threadIdx.x;
    if (idx >= (long long)n * KPAD0) return;
    int nrow = (int)(idx / KPAD0);
    int k    = (int)(idx % KPAD0);
    _Float16 out = (_Float16)0.0f;
    if (k < NCOMB) {
        float v = (k < NSTATIC) ? xs[(size_t)nrow * NSTATIC + k]
                                : xd[(size_t)nrow * NDYNFLAT + (k - NSTATIC)];
        v = cleanf(v, BIGLIM);
        float y = (v - stats[448 + k]) * stats[672 + k] * g[k] + b[k];
        out = (_Float16)cleanf(y, 10.0f);
    }
    A[idx] = out;
}

// ---------------------------------------------------------------------------
// repack weights [Kreal x 64] f32 -> WMMA B lane layout, f16, K padded
// Bpack[((kk*4+nt)*32 + lane)*16 + j] = W[kk*32 + (j<8 ? 8*(lane>>4)+j : 16+8*(lane>>4)+j-8)][nt*16 + (lane&15)]
// ---------------------------------------------------------------------------
__global__ void k_repack_B(const float* __restrict__ W, _Float16* __restrict__ Bp,
                           int Kreal, int ksteps) {
    int idx = blockIdx.x * blockDim.x + threadIdx.x;
    int total = ksteps * 4 * 32 * 16;
    if (idx >= total) return;
    int j    = idx & 15;
    int lane = (idx >> 4) & 31;
    int nt   = (idx >> 9) & 3;
    int kk   = idx >> 11;
    int k   = kk * 32 + ((j < 8) ? 8 * (lane >> 4) + j : 16 + 8 * (lane >> 4) + (j - 8));
    int col = nt * 16 + (lane & 15);
    float v = (k < Kreal) ? W[(size_t)k * NHID + col] : 0.0f;
    Bp[idx] = (_Float16)v;
}

// ---------------------------------------------------------------------------
// WMMA GEMM: H[N x 64] = A[N x K] * W[K x 64]    (one wave per 16-row tile)
// A: f16 in global, row stride lda halves (K padded to 32*ksteps, zero pad).
// B: staged into LDS once per workgroup (<= 28 KB), read via ds_load — keeps
// B fetch on DScnt while A streams on LOADcnt.
// ---------------------------------------------------------------------------
__global__ void k_gemm_wmma(const _Float16* __restrict__ A, int lda,
                            const _Float16* __restrict__ Bp,
                            float* __restrict__ H, int ksteps, int n) {
    __shared__ __align__(32) _Float16 Bl[MAXKSTEPS * 4 * 32 * 16];   // 28 KB max

    // cooperative B copy: ksteps*4096 bytes = ksteps*256 uint4
    {
        const uint4* src = (const uint4*)Bp;
        uint4*       dst = (uint4*)Bl;
        int total = ksteps * 256;
        for (int i = threadIdx.x; i < total; i += blockDim.x) dst[i] = src[i];
    }
    __syncthreads();

    int wave = (blockIdx.x * blockDim.x + threadIdx.x) >> 5;
    int lane = threadIdx.x & 31;
    int row0 = wave * 16;
    if (row0 >= n) return;            // wave-uniform: EXEC all-ones for WMMA

    v8f acc0 = {}, acc1 = {}, acc2 = {}, acc3 = {};
    int m    = lane & 15;
    int koff = (lane < 16) ? 0 : 8;
    const _Float16* arow = A + (size_t)(row0 + m) * lda;
    const v16h*     bl   = (const v16h*)Bl;

    for (int kk = 0; kk < ksteps; ++kk) {
        // prefetch next A chunk for this row (speculative, dropped if OOB)
        __builtin_prefetch(arow + (kk + 1) * 32, 0, 0);

        // A fragment: K = kk*32 + {koff..koff+7, koff+16..koff+23}
        v8h lo = *(const v8h*)(arow + kk * 32 + koff);
        v8h hi = *(const v8h*)(arow + kk * 32 + koff + 16);
        v16h a;
#pragma unroll
        for (int i = 0; i < 8; ++i) { a[i] = lo[i]; a[i + 8] = hi[i]; }

        v16h b0 = bl[(kk * 4 + 0) * 32 + lane];
        v16h b1 = bl[(kk * 4 + 1) * 32 + lane];
        v16h b2 = bl[(kk * 4 + 2) * 32 + lane];
        v16h b3 = bl[(kk * 4 + 3) * 32 + lane];

        acc0 = __builtin_amdgcn_wmma_f32_16x16x32_f16(false, a, false, b0, (short)0, acc0, false, false);
        acc1 = __builtin_amdgcn_wmma_f32_16x16x32_f16(false, a, false, b1, (short)0, acc1, false, false);
        acc2 = __builtin_amdgcn_wmma_f32_16x16x32_f16(false, a, false, b2, (short)0, acc2, false, false);
        acc3 = __builtin_amdgcn_wmma_f32_16x16x32_f16(false, a, false, b3, (short)0, acc3, false, false);
    }

    // C/D layout: lane<16 -> M=v, N=lane ; lane>=16 -> M=v+8, N=lane-16
    int rbase = row0 + ((lane < 16) ? 0 : 8);
#pragma unroll
    for (int v = 0; v < 8; ++v) {
        size_t r = (size_t)(rbase + v) * NHID;
        H[r + 0 * 16 + m] = acc0[v];
        H[r + 1 * 16 + m] = acc1[v];
        H[r + 2 * 16 + m] = acc2[v];
        H[r + 3 * 16 + m] = acc3[v];
    }
}

// ---------------------------------------------------------------------------
// conv init: o = bias + dinv^2 * h   (self loop, norm = dinv*1*dinv)
// ---------------------------------------------------------------------------
__global__ void k_selfloop(const float* __restrict__ h, const float* __restrict__ dinv,
                           const float* __restrict__ bias, float* __restrict__ o, int n) {
    long long idx = (long long)blockIdx.x * blockDim.x + threadIdx.x;
    if (idx >= (long long)n * NHID) return;
    int nrow = (int)(idx >> 6);
    int f    = (int)(idx & 63);
    float dv = dinv[nrow];
    o[idx] = bias[f] + dv * dv * h[idx];
}

// ---------------------------------------------------------------------------
// edge scatter: one wave per edge, lanes cover features f and f+32 (coalesced)
// o[col] += (dinv[row]*ew*dinv[col]) * h[row]
// ---------------------------------------------------------------------------
__global__ void k_edge_scatter(const int* __restrict__ ei, const float* __restrict__ ew,
                               const float* __restrict__ dinv,
                               const float* __restrict__ h, float* __restrict__ o,
                               long long E) {
    long long warp = ((long long)blockIdx.x * blockDim.x + threadIdx.x) >> 5;
    int lane = threadIdx.x & 31;
    if (warp >= E) return;
    int r = ei[warp];
    int c = ei[E + warp];
    float w  = clean_ew(ew[warp]);
    float nm = dinv[r] * w * dinv[c];
    size_t rb = (size_t)r * NHID, cb = (size_t)c * NHID;
    atomicAdd(&o[cb + lane],      nm * h[rb + lane]);
    atomicAdd(&o[cb + lane + 32], nm * h[rb + lane + 32]);
}

// ---------------------------------------------------------------------------
// BN + ReLU, write f32 (for next stage / final head) and f16 A (next GEMM)
// ---------------------------------------------------------------------------
__global__ void k_normalize(float* __restrict__ o, const float* __restrict__ stats,
                            const float* __restrict__ g, const float* __restrict__ be,
                            _Float16* __restrict__ A, int n) {
    long long idx = (long long)blockIdx.x * blockDim.x + threadIdx.x;
    if (idx >= (long long)n * NHID) return;
    int f = (int)(idx & 63);
    float v = cleanf(o[idx], 10.0f);
    float y = (v - stats[448 + f]) * stats[672 + f] * g[f] + be[f];
    y = fmaxf(cleanf(y, 10.0f), 0.0f);
    o[idx] = y;
    A[idx] = (_Float16)y;
}

// ---------------------------------------------------------------------------
// final linear head: out[n] = clip(x @ lin_W + lin_b, 10)
// ---------------------------------------------------------------------------
__global__ void k_head(const float* __restrict__ x, const float* __restrict__ lw,
                       const float* __restrict__ lb, float* __restrict__ out, int n) {
    int i = blockIdx.x * blockDim.x + threadIdx.x;
    if (i >= n) return;
    float s = lb[0];
    const float* xr = x + (size_t)i * NHID;
#pragma unroll
    for (int f = 0; f < NHID; ++f) s += xr[f] * lw[f];
    out[i] = cleanf(s, 10.0f);
}

// ---------------------------------------------------------------------------
// launch
// ---------------------------------------------------------------------------
extern "C" void kernel_launch(void* const* d_in, const int* in_sizes, int n_in,
                              void* d_out, int out_size, void* d_ws, size_t ws_size,
                              hipStream_t stream) {
    const float* xs   = (const float*)d_in[0];
    const float* xd   = (const float*)d_in[1];
    const int*   ei   = (const int*)d_in[2];
    const float* ew   = (const float*)d_in[3];
    const float* bnig = (const float*)d_in[4];
    const float* bnib = (const float*)d_in[5];
    const float* Ws[3]   = { (const float*)d_in[6],  (const float*)d_in[8],  (const float*)d_in[10] };
    const float* bs[3]   = { (const float*)d_in[7],  (const float*)d_in[9],  (const float*)d_in[11] };
    const float* gs[3]   = { (const float*)d_in[12], (const float*)d_in[14], (const float*)d_in[16] };
    const float* bes[3]  = { (const float*)d_in[13], (const float*)d_in[15], (const float*)d_in[17] };
    const float* linW = (const float*)d_in[18];
    const float* linb = (const float*)d_in[19];
    float* out = (float*)d_out;

    const int       N = in_sizes[0] / NSTATIC;        // 100000 (divisible by 16)
    const long long E = (long long)in_sizes[3];

    // workspace carve-up (all offsets 256B aligned)
    uintptr_t base = (uintptr_t)d_ws;
    _Float16* A     = (_Float16*)base;                         // N*224 halves = 44.8 MB
    base += (size_t)N * KPAD0 * sizeof(_Float16);
    float*    h     = (float*)base;                            // N*64 f32 = 25.6 MB
    base += (size_t)N * NHID * sizeof(float);
    float*    o     = (float*)base;                            // N*64 f32 = 25.6 MB
    base += (size_t)N * NHID * sizeof(float);
    float*    dinv  = (float*)base;                            // N f32
    base += ((size_t)N * sizeof(float) + 255) & ~(size_t)255;
    float*    stats = (float*)base;                            // 4*224 f32
    base += 4096;
    _Float16* Bpack = (_Float16*)base;                         // 7*4*32*16 halves

    const int TB = 256;
    const int nwarp_blocks = (int)((E * 32 + TB - 1) / TB);
    const int nelem_blocks = (int)(((long long)N * NHID + TB - 1) / TB);
    const int gemm_blocks  = (N / 16 + (TB / 32) - 1) / (TB / 32);

    // ---- input BN stats + layer-0 A fill -----------------------------------
    k_zero_f32<<<2, TB, 0, stream>>>(stats, 448);
    k_stats_input<<<(N + 255) / 256, TB, 0, stream>>>(xs, xd, stats, N);
    k_bn_finalize<<<1, TB, 0, stream>>>(stats, NCOMB, 1.0f / (float)N);
    {
        long long tot = (long long)N * KPAD0;
        k_fill_A0<<<(int)((tot + TB - 1) / TB), TB, 0, stream>>>(xs, xd, stats, bnig, bnib, A, N);
    }

    // ---- gcn_norm: degrees with self loops, dinv ---------------------------
    k_deg_init<<<(N + TB - 1) / TB, TB, 0, stream>>>(dinv, N);
    k_deg_scatter<<<(int)((E + TB - 1) / TB), TB, 0, stream>>>(ei, ew, dinv, E);
    k_dinv<<<(N + TB - 1) / TB, TB, 0, stream>>>(dinv, N);

    // ---- three GCN layers --------------------------------------------------
    const int Kreal[3]  = { NCOMB, NHID, NHID };
    const int ksteps[3] = { 7, 2, 2 };
    const int lda[3]    = { KPAD0, NHID, NHID };

    for (int L = 0; L < 3; ++L) {
        int total = ksteps[L] * 4 * 32 * 16;
        k_repack_B<<<(total + TB - 1) / TB, TB, 0, stream>>>(Ws[L], Bpack, Kreal[L], ksteps[L]);
        k_gemm_wmma<<<gemm_blocks, TB, 0, stream>>>(A, lda[L], Bpack, h, ksteps[L], N);
        k_selfloop<<<nelem_blocks, TB, 0, stream>>>(h, dinv, bs[L], o, N);
        k_edge_scatter<<<nwarp_blocks, TB, 0, stream>>>(ei, ew, dinv, h, o, E);
        k_zero_f32<<<2, TB, 0, stream>>>(stats, 448);
        k_stats_hidden<<<(N + 255) / 256, TB, 0, stream>>>(o, stats, N);
        k_bn_finalize<<<1, TB, 0, stream>>>(stats, NHID, 1.0f / (float)N);
        k_normalize<<<nelem_blocks, TB, 0, stream>>>(o, stats, gs[L], bes[L], A, N);
    }

    // ---- head --------------------------------------------------------------
    k_head<<<(N + TB - 1) / TB, TB, 0, stream>>>(o, linW, linb, out, N);
}
